// MultiHeadAttentionPooling_45088566673554
// MI455X (gfx1250) — matrix operand order repaired and checked by reading, not compiled, not run on hardware
//
#include <hip/hip_runtime.h>
#include <hip/hip_bf16.h>
#include <math.h>

// Problem constants
#define BB 8
#define SS 1024
#define HH 768
#define NHH 12
#define DHH 64

typedef __attribute__((ext_vector_type(16))) _Float16 v16h;
typedef __attribute__((ext_vector_type(8)))  _Float16 v8h;
typedef __attribute__((ext_vector_type(8)))  float    v8f;

__device__ __forceinline__ v8f wmma_f16(v16h a, v16h b, v8f c) {
    return __builtin_amdgcn_wmma_f32_16x16x32_f16(false, a, false, b, (short)0, c, false, false);
}

__device__ __forceinline__ v8f zero8() {
    v8f z;
#pragma unroll
    for (int i = 0; i < 8; ++i) z[i] = 0.0f;
    return z;
}

// A-fragment (16x32, f16). rowptr points at k=0 of this lane's row (M = lo).
// Layout: half j -> k = hi*16 + j (matches B-fragment order).
__device__ __forceinline__ v16h load_a_frag(const _Float16* rowptr, int hi) {
    return *(const v16h*)(rowptr + hi * 16);
}

// B-fragment (32x16, f16). colptr points at k=0 of this lane's column (N = lo).
// Layout: half j -> k = hi*16 + j (16 contiguous halves = one 32B run).
__device__ __forceinline__ v16h load_b_frag(const _Float16* colptr, int hi) {
    return *(const v16h*)(colptr + hi * 16);
}

// ---------------- prep kernels ----------------
__global__ void mhap_cvt_f32_to_f16(const float* __restrict__ src,
                                    _Float16* __restrict__ dst, int n) {
    int i = blockIdx.x * blockDim.x + threadIdx.x;
    if (i < n) dst[i] = (_Float16)src[i];
}

// hs (b, s, h) f32 -> hsT (b, h, s) f16 ; LDS-tiled 32x32 for coalescing.
__global__ __launch_bounds__(256) void mhap_transpose_hs(
    const float* __restrict__ hs, _Float16* __restrict__ hsT) {
    __shared__ _Float16 tile[32][34];   // pad -> stride 68B, gcd(17,64)=1
    int hx = blockIdx.x % (HH / 32);
    int sx = (blockIdx.x / (HH / 32)) % (SS / 32);
    int b  = blockIdx.x / ((HH / 32) * (SS / 32));
    int tx = threadIdx.x & 31, ty = threadIdx.x >> 5;   // 32 x 8
#pragma unroll
    for (int k = 0; k < 4; ++k) {
        int s = sx * 32 + ty + k * 8;
        tile[ty + k * 8][tx] = (_Float16)hs[((size_t)b * SS + s) * HH + hx * 32 + tx];
    }
    __syncthreads();
#pragma unroll
    for (int k = 0; k < 4; ++k) {
        int h = hx * 32 + ty + k * 8;
        hsT[((size_t)b * HH + h) * SS + sx * 32 + tx] = tile[tx][ty + k * 8];
    }
}

// ---------------- fused Q/K projection: [8192x768] @ W^T + bias ----------------
// One wave computes a 16(M) x 64(N) tile of BOTH Q and K; k-loop is software-
// pipelined and unrolled by 2 (SSA ping-pong buffers -> no register copies).
// Base pointers stay loop-invariant (offsets derived from the scalar counter)
// so address-space inference keeps every load a global_load, not flat_load.
__global__ __launch_bounds__(128) void mhap_qk_proj(
    const _Float16* __restrict__ hs16,
    const _Float16* __restrict__ qw16, const _Float16* __restrict__ kw16,
    const float* __restrict__ qb, const float* __restrict__ kb,
    _Float16* __restrict__ Qh, _Float16* __restrict__ Kh) {
    int gwave = (blockIdx.x * 128 + threadIdx.x) >> 5;
    int lane = threadIdx.x & 31, lo = lane & 15, hi = lane >> 4;
    int mt = gwave / 12;        // 512 M-tiles
    int ng = gwave % 12;        // 12 groups of 64 columns
    size_t row = (size_t)mt * 16 + lo;

    const _Float16* arow = hs16 + row * HH;
    const _Float16* qcol[4];
    const _Float16* kcol[4];
#pragma unroll
    for (int nt = 0; nt < 4; ++nt) {
        int n = ng * 64 + nt * 16 + lo;
        qcol[nt] = qw16 + (size_t)n * HH;
        kcol[nt] = kw16 + (size_t)n * HH;
    }

    v8f cq[4], ck[4];
#pragma unroll
    for (int i = 0; i < 4; ++i) { cq[i] = zero8(); ck[i] = zero8(); }

    v16h a0 = load_a_frag(arow, hi);
    v16h bq0[4], bk0[4];
#pragma unroll
    for (int nt = 0; nt < 4; ++nt) {
        bq0[nt] = load_b_frag(qcol[nt], hi);
        bk0[nt] = load_b_frag(kcol[nt], hi);
    }

    for (int kc = 0; kc < HH / 32; kc += 2) {
        const int o1 = (kc + 1) * 32;
        const int o2 = ((kc + 2 < HH / 32) ? kc + 2 : kc + 1) * 32;
        // stage A: load k-step kc+1, compute kc
        v16h a1 = load_a_frag(arow + o1, hi);
        v16h bq1[4], bk1[4];
#pragma unroll
        for (int nt = 0; nt < 4; ++nt) {
            bq1[nt] = load_b_frag(qcol[nt] + o1, hi);
            bk1[nt] = load_b_frag(kcol[nt] + o1, hi);
        }
#pragma unroll
        for (int nt = 0; nt < 4; ++nt) {
            cq[nt] = wmma_f16(a0, bq0[nt], cq[nt]);
            ck[nt] = wmma_f16(a0, bk0[nt], ck[nt]);
        }
        // stage B: load k-step kc+2 (redundant reload on last iter), compute kc+1
        a0 = load_a_frag(arow + o2, hi);
#pragma unroll
        for (int nt = 0; nt < 4; ++nt) {
            bq0[nt] = load_b_frag(qcol[nt] + o2, hi);
            bk0[nt] = load_b_frag(kcol[nt] + o2, hi);
        }
#pragma unroll
        for (int nt = 0; nt < 4; ++nt) {
            cq[nt] = wmma_f16(a1, bq1[nt], cq[nt]);
            ck[nt] = wmma_f16(a1, bk1[nt], ck[nt]);
        }
    }
#pragma unroll
    for (int nt = 0; nt < 4; ++nt) {
        int n = ng * 64 + nt * 16 + lo;
        float biasq = qb[n], biask = kb[n];
#pragma unroll
        for (int r = 0; r < 8; ++r) {
            size_t m = (size_t)mt * 16 + r + 8 * hi;
            Qh[m * HH + n] = (_Float16)(cq[nt][r] + biasq);
            Kh[m * HH + n] = (_Float16)(ck[nt][r] + biask);
        }
    }
}

// ---------------- attention + head-mean + pooling GEMM ----------------
// One workgroup (4 waves) per (b, 16-query tile). Wave w owns key columns
// [w*256, w*256+256). Pass 1 computes scores once (WMMA), stashes them in a
// 32KB LDS buffer and tracks online max/sum; pass 2 reads LDS, exponentiates
// and accumulates prob/NH into the 32KB f16 mean-weights tile; then the
// pooling GEMM runs straight out of LDS A-fragments. ~66KB LDS (<320KB/WGP).
__global__ __launch_bounds__(128) void mhap_attn(
    const _Float16* __restrict__ Qh, const _Float16* __restrict__ Kh,
    const _Float16* __restrict__ hsT, const int* __restrict__ mask,
    _Float16* __restrict__ pooledh) {
    __shared__ _Float16 wacc[16 * SS];      // 32 KB: mean-over-heads weights
    __shared__ _Float16 sbuf[16 * SS];      // 32 KB: this head's scores
    __shared__ float red[2][4][16];         // [max|sum][wave][row]
    __shared__ float rowstat[2][16];        // final row max / sum

    int b = blockIdx.x >> 6;
    int q0 = (blockIdx.x & 63) * 16;
    int tid = threadIdx.x;
    int wave = tid >> 5, lane = tid & 31, lo = lane & 15, hi = lane >> 4;

    for (int i = tid; i < 16 * SS; i += 128) wacc[i] = (_Float16)0.0f;
    __syncthreads();

    const float scale = 0.125f;             // 1/sqrt(64)
    const int nt0 = wave * 16;
    const _Float16* kwavebase =
        Kh + ((size_t)b * SS + nt0 * 16 + lo) * HH;     // this lane's key rows

    for (int h = 0; h < NHH; ++h) {
        const _Float16* qbase = Qh + ((size_t)b * SS + q0 + lo) * HH + h * DHH;
        v16h a0 = load_a_frag(qbase, hi);
        v16h a1 = load_a_frag(qbase + 32, hi);

        float vmax[8], vsum[8];
#pragma unroll
        for (int r = 0; r < 8; ++r) { vmax[r] = -3.0e38f; vsum[r] = 0.0f; }

        // pass 1: scores -> sbuf ; online row-max / row-sum.
        // Unrolled by 2 with ping-pong fragment buffers; khead stays invariant
        // so loads remain global_load with offsets from the scalar counter.
        const _Float16* khead = kwavebase + h * DHH;
        v16h b0c = load_b_frag(khead, hi);
        v16h b1c = load_b_frag(khead + 32, hi);
        for (int nt = 0; nt < 16; nt += 2) {
            size_t o1 = (size_t)(nt + 1) * 16 * HH;
            size_t o2 = (size_t)((nt + 2 < 16) ? nt + 2 : nt + 1) * 16 * HH;
            // stage A: load nt+1, compute nt
            v16h b0n = load_b_frag(khead + o1, hi);
            v16h b1n = load_b_frag(khead + o1 + 32, hi);
            if (h + 1 < NHH) {
                __builtin_prefetch(khead + (size_t)nt * 16 * HH + DHH, 0, 0);
                __builtin_prefetch(khead + o1 + DHH, 0, 0);
            }
            v8f c = zero8();
            c = wmma_f16(a0, b0c, c);
            c = wmma_f16(a1, b1c, c);
            {
                int n = (nt0 + nt) * 16 + lo;
                bool valid = (mask[b * SS + n] != 0);
#pragma unroll
                for (int r = 0; r < 8; ++r) {
                    float sv = valid ? c[r] * scale : -10000.0f;
                    sbuf[(r + 8 * hi) * SS + n] = (_Float16)sv;
                    float mn = fmaxf(vmax[r], sv);
                    vsum[r] = vsum[r] * __expf(vmax[r] - mn) + __expf(sv - mn);
                    vmax[r] = mn;
                }
            }
            // stage B: load nt+2 (redundant reload on last iter), compute nt+1
            b0c = load_b_frag(khead + o2, hi);
            b1c = load_b_frag(khead + o2 + 32, hi);
            v8f c2 = zero8();
            c2 = wmma_f16(a0, b0n, c2);
            c2 = wmma_f16(a1, b1n, c2);
            {
                int n = (nt0 + nt + 1) * 16 + lo;
                bool valid = (mask[b * SS + n] != 0);
#pragma unroll
                for (int r = 0; r < 8; ++r) {
                    float sv = valid ? c2[r] * scale : -10000.0f;
                    sbuf[(r + 8 * hi) * SS + n] = (_Float16)sv;
                    float mn = fmaxf(vmax[r], sv);
                    vsum[r] = vsum[r] * __expf(vmax[r] - mn) + __expf(sv - mn);
                    vmax[r] = mn;
                }
            }
        }
        // cross-lane combine within the 16-lane group (same hi -> same rows)
#pragma unroll
        for (int off = 1; off < 16; off <<= 1) {
#pragma unroll
            for (int r = 0; r < 8; ++r) {
                float om = __shfl_xor(vmax[r], off, 32);
                float os = __shfl_xor(vsum[r], off, 32);
                float mn = fmaxf(vmax[r], om);
                vsum[r] = vsum[r] * __expf(vmax[r] - mn) + os * __expf(om - mn);
                vmax[r] = mn;
            }
        }
        if (lo == 0) {
#pragma unroll
            for (int r = 0; r < 8; ++r) {
                red[0][wave][r + 8 * hi] = vmax[r];
                red[1][wave][r + 8 * hi] = vsum[r];
            }
        }
        __syncthreads();
        if (tid < 16) {
            float m = red[0][0][tid], s = red[1][0][tid];
            for (int w = 1; w < 4; ++w) {
                float om = red[0][w][tid], os = red[1][w][tid];
                float mn = fmaxf(m, om);
                s = s * __expf(m - mn) + os * __expf(om - mn);
                m = mn;
            }
            rowstat[0][tid] = m;
            rowstat[1][tid] = s;
        }
        __syncthreads();

        float rmax[8], rinv[8];
#pragma unroll
        for (int r = 0; r < 8; ++r) {
            rmax[r] = rowstat[0][r + 8 * hi];
            rinv[r] = 1.0f / rowstat[1][r + 8 * hi];
        }
        // pass 2: read scores from LDS, accumulate prob/NH into wacc
        for (int nt = 0; nt < 16; ++nt) {
            int n = (nt0 + nt) * 16 + lo;
#pragma unroll
            for (int r = 0; r < 8; ++r) {
                int idx = (r + 8 * hi) * SS + n;
                float sv = (float)sbuf[idx];
                float p = __expf(sv - rmax[r]) * rinv[r] * (1.0f / 12.0f);
                wacc[idx] = (_Float16)((float)wacc[idx] + p);
            }
        }
        __syncthreads();
    }

    // pooled (16 x 768) = wacc (16 x 1024) @ hs[b] (1024 x 768); B from hsT.
    v8f pc[12];
#pragma unroll
    for (int i = 0; i < 12; ++i) pc[i] = zero8();
    for (int kc = 0; kc < SS / 32; ++kc) {
        v16h a = load_a_frag(&wacc[lo * SS + kc * 32], hi);
#pragma unroll
        for (int g = 0; g < 2; ++g) {
            v16h bf[6];
#pragma unroll
            for (int i = 0; i < 6; ++i) {
                int n = (wave * 12 + g * 6 + i) * 16 + lo;
                bf[i] = load_b_frag(hsT + ((size_t)b * HH + n) * SS + kc * 32, hi);
            }
#pragma unroll
            for (int i = 0; i < 6; ++i)
                pc[g * 6 + i] = wmma_f16(a, bf[i], pc[g * 6 + i]);
        }
    }
#pragma unroll
    for (int i = 0; i < 12; ++i) {
        int n = (wave * 12 + i) * 16 + lo;
#pragma unroll
        for (int r = 0; r < 8; ++r) {
            size_t m = (size_t)b * SS + q0 + r + 8 * hi;
            pooledh[m * HH + n] = (_Float16)pc[i][r];
        }
    }
}

// ---------------- output projection: y = pooled @ o_w^T + o_b ----------------
__global__ __launch_bounds__(128) void mhap_oproj(
    const _Float16* __restrict__ ph, const _Float16* __restrict__ ow16,
    const float* __restrict__ ob, float* __restrict__ y) {
    int gwave = (blockIdx.x * 128 + threadIdx.x) >> 5;
    int lane = threadIdx.x & 31, lo = lane & 15, hi = lane >> 4;
    int mt = gwave / 12;
    int ng = gwave % 12;
    size_t row = (size_t)mt * 16 + lo;

    const _Float16* arow = ph + row * HH;
    const _Float16* ocol[4];
#pragma unroll
    for (int nt = 0; nt < 4; ++nt)
        ocol[nt] = ow16 + (size_t)(ng * 64 + nt * 16 + lo) * HH;

    v8f c[4];
#pragma unroll
    for (int i = 0; i < 4; ++i) c[i] = zero8();

    v16h a0 = load_a_frag(arow, hi);
    v16h bf0[4];
#pragma unroll
    for (int nt = 0; nt < 4; ++nt) bf0[nt] = load_b_frag(ocol[nt], hi);

    for (int kc = 0; kc < HH / 32; kc += 2) {
        const int o1 = (kc + 1) * 32;
        const int o2 = ((kc + 2 < HH / 32) ? kc + 2 : kc + 1) * 32;
        v16h a1 = load_a_frag(arow + o1, hi);
        v16h bf1[4];
#pragma unroll
        for (int nt = 0; nt < 4; ++nt) bf1[nt] = load_b_frag(ocol[nt] + o1, hi);
#pragma unroll
        for (int nt = 0; nt < 4; ++nt) c[nt] = wmma_f16(a0, bf0[nt], c[nt]);

        a0 = load_a_frag(arow + o2, hi);
#pragma unroll
        for (int nt = 0; nt < 4; ++nt) bf0[nt] = load_b_frag(ocol[nt] + o2, hi);
#pragma unroll
        for (int nt = 0; nt < 4; ++nt) c[nt] = wmma_f16(a1, bf1[nt], c[nt]);
    }
#pragma unroll
    for (int nt = 0; nt < 4; ++nt) {
        int n = ng * 64 + nt * 16 + lo;
        float bias = ob[n];
#pragma unroll
        for (int r = 0; r < 8; ++r) {
            size_t m = (size_t)mt * 16 + r + 8 * hi;
            y[m * HH + n] = c[nt][r] + bias;
        }
    }
}

// ---------------- LayerNorm over last dim (768) ----------------
__global__ __launch_bounds__(256) void mhap_ln(
    const float* __restrict__ y, const float* __restrict__ lnw,
    const float* __restrict__ lnb, float* __restrict__ out) {
    __shared__ float rs[2][8];
    size_t row = blockIdx.x;
    const float* yr = y + row * HH;
    float s = 0.0f, ss = 0.0f;
    for (int i = threadIdx.x; i < HH; i += 256) {
        float v = yr[i];
        s += v; ss += v * v;
    }
#pragma unroll
    for (int off = 16; off > 0; off >>= 1) {
        s += __shfl_down(s, off, 32);
        ss += __shfl_down(ss, off, 32);
    }
    int wave = threadIdx.x >> 5;
    if ((threadIdx.x & 31) == 0) { rs[0][wave] = s; rs[1][wave] = ss; }
    __syncthreads();
    if (threadIdx.x == 0) {
        float a = 0.0f, b2 = 0.0f;
        for (int w = 0; w < 8; ++w) { a += rs[0][w]; b2 += rs[1][w]; }
        rs[0][0] = a; rs[1][0] = b2;
    }
    __syncthreads();
    float mu = rs[0][0] * (1.0f / HH);
    float var = rs[1][0] * (1.0f / HH) - mu * mu;
    float inv = rsqrtf(var + 1e-5f);
    for (int i = threadIdx.x; i < HH; i += 256) {
        out[row * HH + i] = (yr[i] - mu) * inv * lnw[i] + lnb[i];
    }
}

extern "C" void kernel_launch(void* const* d_in, const int* in_sizes, int n_in,
                              void* d_out, int out_size, void* d_ws, size_t ws_size,
                              hipStream_t stream) {
    const float* hs  = (const float*)d_in[0];
    const int*   msk = (const int*)d_in[1];
    const float* qw  = (const float*)d_in[2];
    const float* qb  = (const float*)d_in[3];
    const float* kw  = (const float*)d_in[4];
    const float* kb  = (const float*)d_in[5];
    // d_in[6], d_in[7] (v_w, v_b) are unused by the reference
    const float* ow  = (const float*)d_in[8];
    const float* ob  = (const float*)d_in[9];
    const float* lnw = (const float*)d_in[10];
    const float* lnb = (const float*)d_in[11];
    float* out = (float*)d_out;

    const size_t BSH = (size_t)BB * SS * HH;      // 6,291,456
    const size_t WW  = (size_t)HH * HH;           // 589,824

    char* ws = (char*)d_ws;
    size_t off = 0;
    auto alloc = [&](size_t bytes) -> void* {
        void* p = ws + off;
        off += (bytes + 255) & ~(size_t)255;
        return p;
    };
    _Float16* hs16    = (_Float16*)alloc(BSH * 2);
    _Float16* hsT16   = (_Float16*)alloc(BSH * 2);
    _Float16* qw16    = (_Float16*)alloc(WW * 2);
    _Float16* kw16    = (_Float16*)alloc(WW * 2);
    _Float16* ow16    = (_Float16*)alloc(WW * 2);
    _Float16* Qh      = (_Float16*)alloc(BSH * 2);
    _Float16* Kh      = (_Float16*)alloc(BSH * 2);
    _Float16* pooledh = (_Float16*)alloc(BSH * 2);
    float*    yb      = (float*)alloc(BSH * 4);

    // prep: f16 conversions + tiled transpose
    mhap_cvt_f32_to_f16<<<(int)((BSH + 255) / 256), 256, 0, stream>>>(hs, hs16, (int)BSH);
    mhap_cvt_f32_to_f16<<<(int)((WW + 255) / 256), 256, 0, stream>>>(qw, qw16, (int)WW);
    mhap_cvt_f32_to_f16<<<(int)((WW + 255) / 256), 256, 0, stream>>>(kw, kw16, (int)WW);
    mhap_cvt_f32_to_f16<<<(int)((WW + 255) / 256), 256, 0, stream>>>(ow, ow16, (int)WW);
    mhap_transpose_hs<<<BB * (SS / 32) * (HH / 32), 256, 0, stream>>>(hs, hsT16);

    // fused Q/K projection: 512 M-tiles * 12 N-groups = 6144 waves / 4 per block
    mhap_qk_proj<<<1536, 128, 0, stream>>>(hs16, qw16, kw16, qb, kb, Qh, Kh);

    // attention + head-mean + pooling GEMM: one block per (b, 16-query tile)
    mhap_attn<<<BB * (SS / 16), 128, 0, stream>>>(Qh, Kh, hsT16, msk, pooledh);

    // output projection
    mhap_oproj<<<1536, 128, 0, stream>>>(pooledh, ow16, ob, yb);

    // layernorm
    mhap_ln<<<BB * SS, 256, 0, stream>>>(yb, lnw, lnb, out);
}
